// Model_21981642621060
// MI455X (gfx1250) — compile-verified, matrix operand order checked by reading, not run on hardware
//
#include <hip/hip_runtime.h>
#include <hip/hip_bf16.h>

// ---------------------------------------------------------------------------
// Model dims (fixed): B=32, T=128, E=512, H=256, V=32000, 4H=1024, M=B*T=4096
// ---------------------------------------------------------------------------

typedef __bf16 bf16_t;
typedef bf16_t bfvec16 __attribute__((ext_vector_type(16)));
typedef float  fvec8   __attribute__((ext_vector_type(8)));

union Frag {
    uint4   q[2];
    bfvec16 v;
};

static __device__ __forceinline__ unsigned short f2bf(float x) {
    union { float f; unsigned u; } v;
    v.f = x;
    unsigned r = (v.u + 0x7FFFu + ((v.u >> 16) & 1u)) >> 16;
    return (unsigned short)r;
}

static __device__ __forceinline__ fvec8 zero8() {
    fvec8 z = {0.f, 0.f, 0.f, 0.f, 0.f, 0.f, 0.f, 0.f};
    return z;
}

// Branch-free fast activations built on native v_exp_f32 / v_rcp_f32.
static __device__ __forceinline__ float sigmoid_fast(float x) {
    return __builtin_amdgcn_rcpf(1.0f + __expf(-x));
}
static __device__ __forceinline__ float tanh_fast(float x) {
    const float e = __expf(-2.0f * __builtin_fabsf(x));
    const float t = (1.0f - e) * __builtin_amdgcn_rcpf(1.0f + e);
    return __builtin_copysignf(t, x);
}

// ---------------------------------------------------------------------------
// Embedding gather -> bf16 rows, row m = t*32 + b
// ---------------------------------------------------------------------------
__global__ void gather_embed_bf16(const int* __restrict__ x_ind,
                                  const float* __restrict__ embed,
                                  unsigned short* __restrict__ Xbf) {
    const int m = blockIdx.x;           // 0..4095 ; m = t*32 + b
    const int t = m >> 5;
    const int b = m & 31;
    const int row = x_ind[b * 128 + t]; // x_ind is [B, T]
    const float* src = embed + (size_t)row * 512;
    unsigned short* dst = Xbf + (size_t)m * 512;
    for (int e = threadIdx.x; e < 512; e += blockDim.x)
        dst[e] = f2bf(src[e]);
}

// ---------------------------------------------------------------------------
// Tiled transpose + fp32 -> bf16 convert:  src[R][C] -> dst[C][R]
// grid = (C/32, R/32), block = (32, 8)
// ---------------------------------------------------------------------------
__global__ void transpose_f32_to_bf16(const float* __restrict__ src,
                                      unsigned short* __restrict__ dst,
                                      int R, int C) {
    __shared__ float tile[32][33];
    const int c0 = blockIdx.x * 32;
    const int r0 = blockIdx.y * 32;
    for (int i = threadIdx.y; i < 32; i += 8)
        tile[i][threadIdx.x] = src[(size_t)(r0 + i) * C + (c0 + threadIdx.x)];
    __syncthreads();
    for (int i = threadIdx.y; i < 32; i += 8)
        dst[(size_t)(c0 + i) * R + (r0 + threadIdx.x)] = f2bf(tile[threadIdx.x][i]);
}

// ---------------------------------------------------------------------------
// Concatenate 4 gate biases into one [1024] vector
// ---------------------------------------------------------------------------
__global__ void pack_bias(const float* __restrict__ bi, const float* __restrict__ bf,
                          const float* __restrict__ bg, const float* __restrict__ bo,
                          float* __restrict__ bcat) {
    const int i = threadIdx.x;  // 1024 threads
    const float* p = (i < 256) ? bi : (i < 512) ? bf : (i < 768) ? bg : bo;
    bcat[i] = p[i & 255];
}

// ---------------------------------------------------------------------------
// bf16 GEMM:  out[M][N] = A[M][K] * BT[N][K]^T + bias[N]
// Block tile 128x128, 256 threads = 8 waves (wave32), wave tile 32x64.
// LDS staging with padded stride (40 halves) to avoid bank conflicts.
// Next-K-tile rows are prefetched (global_prefetch_b8) while WMMAs run.
// REMAP=1: row m = t*32+b is written to out[b*128 + t] (logits [B,T,V])
// with non-temporal stores (output is write-once; keeps A/B hot in L2).
// ---------------------------------------------------------------------------
template <int REMAP>
__global__ __launch_bounds__(256) void gemm_bf16(const unsigned short* __restrict__ A,
                                                 const unsigned short* __restrict__ BT,
                                                 const float* __restrict__ bias,
                                                 float* __restrict__ out,
                                                 int M, int N, int K) {
    constexpr int LDT = 40;  // LDS row stride in halves (80B, 16B aligned)
    __shared__ unsigned short As[128 * LDT];
    __shared__ unsigned short Bs[128 * LDT];

    const int tid  = threadIdx.x;
    const int lane = tid & 31;
    const int wave = tid >> 5;
    const int bn = blockIdx.x * 128;
    const int bm = blockIdx.y * 128;
    const int wm = (wave >> 1) * 32;   // 0,32,64,96
    const int wn = (wave & 1) * 64;    // 0,64

    const int lrow = lane & 15;
    const int lhi  = lane >> 4;        // 0/1
    const int lk0  = lhi * 8;          // K half-offset 0 or 8

    fvec8 acc[2][4];
#pragma unroll
    for (int i = 0; i < 2; ++i)
#pragma unroll
        for (int j = 0; j < 4; ++j) acc[i][j] = zero8();

    for (int kt = 0; kt < K; kt += 32) {
        // Stage A/B tiles: 128 rows x 32 halves each; 512 uint4 loads per tile.
#pragma unroll
        for (int l = 0; l < 2; ++l) {
            const int idx = tid * 2 + l;       // 0..511
            const int row = idx >> 2;
            const int seg = idx & 3;
            const uint4 av = *(const uint4*)(A  + (size_t)(bm + row) * K + kt + seg * 8);
            const uint4 bv = *(const uint4*)(BT + (size_t)(bn + row) * K + kt + seg * 8);
            *(uint4*)(As + row * LDT + seg * 8) = av;
            *(uint4*)(Bs + row * LDT + seg * 8) = bv;
        }
        // Prefetch next K tile (one cacheline per row; threads 0-127 cover A
        // rows, 128-255 cover B rows).
        if (kt + 32 < K) {
            const int prow = tid & 127;
            const unsigned short* p =
                (tid < 128) ? (A  + (size_t)(bm + prow) * K + kt + 32)
                            : (BT + (size_t)(bn + prow) * K + kt + 32);
            __builtin_prefetch(p, 0, 0);
        }
        __syncthreads();

        Frag a[2], b[4];
#pragma unroll
        for (int i = 0; i < 2; ++i) {
            const int r = wm + i * 16 + lrow;
            a[i].q[0] = *(const uint4*)(As + r * LDT + lk0);
            a[i].q[1] = *(const uint4*)(As + r * LDT + lk0 + 16);
        }
#pragma unroll
        for (int j = 0; j < 4; ++j) {
            const int r = wn + j * 16 + lrow;
            b[j].q[0] = *(const uint4*)(Bs + r * LDT + lk0);
            b[j].q[1] = *(const uint4*)(Bs + r * LDT + lk0 + 16);
        }
#pragma unroll
        for (int i = 0; i < 2; ++i)
#pragma unroll
            for (int j = 0; j < 4; ++j)
                acc[i][j] = __builtin_amdgcn_wmma_f32_16x16x32_bf16(
                    false, a[i].v, false, b[j].v, (short)0, acc[i][j], false, false);
        __syncthreads();
    }

    // Writeback. C/D layout: VGPR r -> M = r + 8*(lane>=16), N = lane&15.
#pragma unroll
    for (int i = 0; i < 2; ++i) {
#pragma unroll
        for (int j = 0; j < 4; ++j) {
#pragma unroll
            for (int r = 0; r < 8; ++r) {
                const int m = bm + wm + i * 16 + r + 8 * lhi;
                const int n = bn + wn + j * 16 + lrow;
                const float v = acc[i][j][r] + bias[n];
                if (REMAP) {
                    // m = t*32 + b  ->  out[b][t][n], T=128. Write-once output:
                    // non-temporal store keeps the reused GEMM operands in L2.
                    const int bb = m & 31;
                    const int tt = m >> 5;
                    __builtin_nontemporal_store(
                        v, &out[((size_t)bb * 128 + tt) * (size_t)N + n]);
                } else {
                    out[(size_t)m * N + n] = v;
                }
            }
        }
    }
}

// ---------------------------------------------------------------------------
// LSTM recurrence: one persistent workgroup, 512 threads = 16 waves.
// Wave w owns h-columns j0..j0+15 (j0 = w*16) for ALL four gates and BOTH
// 16-row batch tiles -> each weight fragment feeds two WMMAs, and the gate
// math plus cell state c stay entirely in registers. h ping-pongs through
// padded LDS; one barrier per step. A per-iteration opaque 32-bit offset
// defeats LICM of the t-invariant weight fragments (which would otherwise
// be hoisted and spilled) while keeping the base pointer a recognized
// global pointer so loads lower as global_load (LOADcnt only), not flat.
// K-loop unrolled x2 so next k-step's weight loads issue under the current
// WMMAs (the recurrence is the serial, latency-critical part).
// xproj: [T*32, 1024] fp32 (bias pre-added). WhT: [1024][256] bf16.
// hs out: [T*32, 256] bf16 for the logits GEMM.
// ---------------------------------------------------------------------------
__global__ __launch_bounds__(512) void lstm_scan(const float* __restrict__ xproj,
                                                 const unsigned short* __restrict__ WhT,
                                                 unsigned short* __restrict__ hs) {
    constexpr int HLD = 264;  // padded h row stride in halves (528B)
    __shared__ unsigned short hbuf[2][32 * HLD];

    const int tid  = threadIdx.x;
    const int lane = tid & 31;
    const int wave = tid >> 5;         // 0..15
    const int j0   = wave * 16;        // h column block
    const int lrow = lane & 15;
    const int lhi  = lane >> 4;
    const int lk0  = lhi * 8;

    // h_0 = 0
    for (int i = tid; i < 32 * HLD; i += 512) hbuf[0][i] = 0;
    __syncthreads();

    fvec8 cacc[2] = {zero8(), zero8()};   // cell state, batch tiles 0 / 1
    int cur = 0;

    for (int t = 0; t < 128; ++t) {
        // Opaque (uniform, actually 0) offset recomputed every iteration:
        // blocks loop-invariant hoisting of the weight fragment loads.
        unsigned whoff = 0;
        asm volatile("" : "+s"(whoff));

        // Seed gate accumulators with xproj (bias already folded in).
        fvec8 z[4][2];
        const float* xp = xproj + (size_t)(t * 32) * 1024;
#pragma unroll
        for (int g = 0; g < 4; ++g) {
#pragma unroll
            for (int mt = 0; mt < 2; ++mt) {
#pragma unroll
                for (int r = 0; r < 8; ++r) {
                    const int bmr = mt * 16 + r + 8 * lhi;
                    z[g][mt][r] = xp[bmr * 1024 + g * 256 + j0 + lrow];
                }
            }
        }

        // z += h_{t-1} @ Wh   (K = 256, 8 k-steps of 32; unroll x2 for
        // load/WMMA overlap while keeping register pressure moderate)
#pragma unroll 2
        for (int kt = 0; kt < 256; kt += 32) {
            Frag a0, a1;
            a0.q[0] = *(const uint4*)(hbuf[cur] + lrow * HLD + kt + lk0);
            a0.q[1] = *(const uint4*)(hbuf[cur] + lrow * HLD + kt + lk0 + 16);
            a1.q[0] = *(const uint4*)(hbuf[cur] + (16 + lrow) * HLD + kt + lk0);
            a1.q[1] = *(const uint4*)(hbuf[cur] + (16 + lrow) * HLD + kt + lk0 + 16);
#pragma unroll
            for (int g = 0; g < 4; ++g) {
                Frag b;
                const unsigned nr = g * 256 + j0 + lrow;
                const unsigned bo0 = whoff + nr * 256 + kt + lk0;
                b.q[0] = *(const uint4*)(WhT + bo0);
                b.q[1] = *(const uint4*)(WhT + bo0 + 16);
                z[g][0] = __builtin_amdgcn_wmma_f32_16x16x32_bf16(
                    false, a0.v, false, b.v, (short)0, z[g][0], false, false);
                z[g][1] = __builtin_amdgcn_wmma_f32_16x16x32_bf16(
                    false, a1.v, false, b.v, (short)0, z[g][1], false, false);
            }
        }

        // Gates + state update, all in registers, branch-free.
        unsigned short* hout = hbuf[cur ^ 1];
        unsigned short* hg   = hs + (size_t)(t * 32) * 256;
#pragma unroll
        for (int mt = 0; mt < 2; ++mt) {
#pragma unroll
            for (int r = 0; r < 8; ++r) {
                const float iv = sigmoid_fast(z[0][mt][r]);
                const float fv = sigmoid_fast(z[1][mt][r]);
                const float gv = tanh_fast(z[2][mt][r]);
                const float ov = sigmoid_fast(z[3][mt][r]);
                const float c  = fv * cacc[mt][r] + iv * gv;
                cacc[mt][r] = c;
                const float h = ov * tanh_fast(c);
                const unsigned short hb = f2bf(h);
                const int bmr = mt * 16 + r + 8 * lhi;
                hout[bmr * HLD + j0 + lrow] = hb;
                hg[bmr * 256 + j0 + lrow]   = hb;
            }
        }
        __syncthreads();
        cur ^= 1;
    }
}

// ---------------------------------------------------------------------------
// Host-side orchestration
// ---------------------------------------------------------------------------
extern "C" void kernel_launch(void* const* d_in, const int* in_sizes, int n_in,
                              void* d_out, int out_size, void* d_ws, size_t ws_size,
                              hipStream_t stream) {
    (void)in_sizes; (void)n_in; (void)out_size; (void)ws_size;

    const int*   x_ind = (const int*)d_in[0];
    const float* embed = (const float*)d_in[1];
    const float* U[4]  = {(const float*)d_in[2], (const float*)d_in[3],
                          (const float*)d_in[4], (const float*)d_in[5]};
    const float* Wg[4] = {(const float*)d_in[6], (const float*)d_in[7],
                          (const float*)d_in[8], (const float*)d_in[9]};
    const float* bi = (const float*)d_in[10];
    const float* bf = (const float*)d_in[11];
    const float* bg = (const float*)d_in[12];
    const float* bo = (const float*)d_in[13];
    const float* W_hy = (const float*)d_in[14];
    const float* b_y  = (const float*)d_in[15];
    float* out = (float*)d_out;

    // Scratch layout
    char* w = (char*)d_ws;
    size_t off = 0;
    auto alloc = [&](size_t bytes) -> void* {
        void* p = w + off;
        off = (off + bytes + 255) & ~(size_t)255;
        return p;
    };
    unsigned short* Xbf  = (unsigned short*)alloc((size_t)4096 * 512 * 2);   //  4 MB
    unsigned short* UT   = (unsigned short*)alloc((size_t)1024 * 512 * 2);   //  1 MB
    unsigned short* WhT  = (unsigned short*)alloc((size_t)1024 * 256 * 2);   // .5 MB
    unsigned short* WT   = (unsigned short*)alloc((size_t)32000 * 256 * 2);  // 16 MB
    float*          bcat = (float*)alloc((size_t)1024 * 4);
    float*          xprj = (float*)alloc((size_t)4096 * 1024 * 4);           // 16 MB
    unsigned short* hsb  = (unsigned short*)alloc((size_t)4096 * 256 * 2);   //  2 MB

    const dim3 tb(32, 8);

    // 1) embed gather -> bf16
    gather_embed_bf16<<<4096, 256, 0, stream>>>(x_ind, embed, Xbf);

    // 2) transpose+convert weights (Bt form: [N][K])
    for (int g = 0; g < 4; ++g)  // U_g: [512,256] -> UT rows g*256.., ld 512
        transpose_f32_to_bf16<<<dim3(8, 16), tb, 0, stream>>>(
            U[g], UT + (size_t)g * 256 * 512, 512, 256);
    for (int g = 0; g < 4; ++g)  // W_g: [256,256] -> WhT rows g*256.., ld 256
        transpose_f32_to_bf16<<<dim3(8, 8), tb, 0, stream>>>(
            Wg[g], WhT + (size_t)g * 256 * 256, 256, 256);
    // W_hy: [256,32000] -> WT [32000][256]
    transpose_f32_to_bf16<<<dim3(1000, 8), tb, 0, stream>>>(W_hy, WT, 256, 32000);

    // 3) bias concat
    pack_bias<<<1, 1024, 0, stream>>>(bi, bf, bg, bo, bcat);

    // 4) xproj = X @ U + b   (M=4096, N=1024, K=512), rows m = t*32+b
    gemm_bf16<0><<<dim3(8, 32), 256, 0, stream>>>(Xbf, UT, bcat, xprj, 4096, 1024, 512);

    // 5) recurrence (single persistent workgroup, 16 waves)
    lstm_scan<<<1, 512, 0, stream>>>(xprj, WhT, hsb);

    // 6) logits = hs @ W_hy + b_y  (M=4096, N=32000, K=256) -> [B,T,V]
    gemm_bf16<1><<<dim3(250, 32), 256, 0, stream>>>(hsb, WT, b_y, out, 4096, 32000, 256);
}
